// Attention_24721831756366
// MI455X (gfx1250) — matrix operand order repaired and checked by reading, not compiled, not run on hardware
//
#include <hip/hip_runtime.h>
#include <hip/hip_bf16.h>

// ---------------------------------------------------------------------------
// MI455X (gfx1250) pipeline:
//   K0a: xh = f16(x)                       elementwise cvt
//   K0b: wqkvT/wprojT = f16(w^T)           cvt + transpose (weights become N x K)
//   K1 : qkv_h  = f16( xh @ w_qkv + b )    WMMA f16 GEMM, TDM tile loads
//   K2 : attn_h = per-position 16x16 attn  WMMA 16x16x64 + 16x16x32(pad)
//   K3 : out    = attn_h @ w_proj + b      WMMA f16 GEMM, TDM tile loads
// Workspace: ~680 MB of f16 staging (assumed available in d_ws).
// ---------------------------------------------------------------------------

typedef __attribute__((ext_vector_type(16))) _Float16 v16h;
typedef __attribute__((ext_vector_type(8)))  _Float16 v8h;
typedef __attribute__((ext_vector_type(8)))  float    v8f;
typedef __attribute__((ext_vector_type(4)))  unsigned int u32x4;
typedef __attribute__((ext_vector_type(8)))  int i32x8;
typedef __attribute__((ext_vector_type(4)))  int i32x4;

union V16 { v16h v; v8h h[2]; };

#define WMMA_F16(a, b, c) \
  __builtin_amdgcn_wmma_f32_16x16x32_f16(false, (a), false, (b), (short)0, (c), false, false)

// Tensor Data Mover availability (probe-verified builtins; arity differs by toolchain)
#if __has_builtin(__builtin_amdgcn_tensor_load_to_lds) && \
    __has_builtin(__builtin_amdgcn_s_wait_tensorcnt)
#define USE_TDM 1
#else
#define USE_TDM 0
#endif

constexpr int BM = 128, BN = 128, BK = 32, TH = 256;
constexpr int AST = BK + 8;   // LDS row stride in halves: 64B data + 16B pad = 80B

#if USE_TDM
// Issue one TDM 2D tile load: nrows=128 rows x 32 halves, row stride K halves,
// into LDS with 16B pad after every 64B row (matches AST layout).
__device__ __forceinline__ void tdm_tile_load(const _Float16* gsrc, void* ldsdst,
                                              int K, int nrows) {
  const unsigned lds = (unsigned)(uintptr_t)ldsdst;            // low 32 = LDS offset
  const unsigned long long ga = (unsigned long long)(uintptr_t)gsrc;
  u32x4 g0;
  g0[0] = 1u;                                                  // count=1, user descriptor
  g0[1] = lds;                                                 // lds_addr (bytes)
  g0[2] = (unsigned)(ga & 0xffffffffu);                        // global_addr[31:0]
  g0[3] = (unsigned)((ga >> 32) & 0x01ffffffu) | 0x80000000u;  // addr[56:32] | type=2
  i32x8 g1;
  g1[0] = (int)((1u << 16) |       // data_size = 2 bytes
                (1u << 20) |       // pad_enable
                (3u << 22) |       // pad_interval: 16 DWORDs (64B = one tile row)
                (3u << 25));       // pad_amount:    4 DWORDs (16B)
  g1[1] = (int)(((unsigned)K & 0xffffu) << 16);                         // tensor_dim0 lo
  g1[2] = (int)((((unsigned)K >> 16) & 0xffffu) |
                (((unsigned)nrows & 0xffffu) << 16));                   // td0 hi | td1 lo
  g1[3] = (int)((((unsigned)nrows >> 16) & 0xffffu) | (32u << 16));     // td1 hi | tile_dim0=32
  g1[4] = (int)(128u);                                                  // tile_dim1=128
  g1[5] = (int)K;                                                       // dim0_stride lo
  g1[6] = 0;
  g1[7] = 0;                                                            // dim1_stride (2D)
  i32x4 g2 = {}; i32x4 g3 = {};
#if defined(__clang_major__) && (__clang_major__ >= 23)
  i32x8 g4 = {};
  __builtin_amdgcn_tensor_load_to_lds(g0, g1, g2, g3, g4, 0);
#else
  __builtin_amdgcn_tensor_load_to_lds(g0, g1, g2, g3, 0);
#endif
}
#endif

// ---------------------------------------------------------------------------
// GEMM: C[M,N] = A[M,K](f16) * BT[N,K](f16, pre-transposed) + bias[N](f32)
// 256 thr = 8 waves; 128x128 tile, BK=32, double-buffered LDS fed by TDM.
// Wave grid 4(M) x 2(N): each wave owns 2x4 WMMA 16x16 accumulators.
// ---------------------------------------------------------------------------
template <bool OUT_HALF>
__global__ __launch_bounds__(256) void gemm_f16_kernel(
    const _Float16* __restrict__ A, const _Float16* __restrict__ BT,
    const float* __restrict__ bias, void* __restrict__ Cpv,
    int M, int N, int K)
{
  __shared__ alignas(16) _Float16 As[2][BM][AST];
  __shared__ alignas(16) _Float16 Bs[2][BN][AST];

  const int tid  = threadIdx.x;
  const int lane = tid & 31;
  const int wave = tid >> 5;
  const int wm   = wave >> 1;       // 0..3  (M)
  const int wn   = wave & 1;        // 0..1  (N)
  const int hs   = lane >> 4;
  const int l16  = lane & 15;

  const int bM = blockIdx.y * BM;
  const int bN = blockIdx.x * BN;
  const int KT = K / BK;

  auto copyTiles = [&](int kt, int buf) {
#if USE_TDM
    if (wave == 0) {   // TDM ignores EXEC; one DMA per descriptor per wave
      tdm_tile_load(A  + (size_t)bM * K + (size_t)kt * BK, &As[buf][0][0], K, M);
      tdm_tile_load(BT + (size_t)bN * K + (size_t)kt * BK, &Bs[buf][0][0], K, N);
    }
#else
#pragma unroll
    for (int it = 0; it < 2; ++it) {
      const int r = (tid >> 2) + it * 64;
      const int c = (tid & 3) * 8;
      *(v8h*)&As[buf][r][c] = *(const v8h*)(A  + (size_t)(bM + r) * K + (size_t)kt * BK + c);
      *(v8h*)&Bs[buf][r][c] = *(const v8h*)(BT + (size_t)(bN + r) * K + (size_t)kt * BK + c);
    }
#endif
  };
  auto waitTiles = [&]() {
#if USE_TDM
    if (wave == 0) __builtin_amdgcn_s_wait_tensorcnt(0);
#endif
  };

  v8f acc[2][4];
#pragma unroll
  for (int i = 0; i < 2; ++i)
#pragma unroll
    for (int j = 0; j < 4; ++j) acc[i][j] = (v8f){};

  copyTiles(0, 0);
  waitTiles();
  __syncthreads();

  int buf = 0;
  for (int kt = 0; kt < KT; ++kt) {
    if (kt + 1 < KT) copyTiles(kt + 1, buf ^ 1);

    // A fragments: 16-bit A 16x32 layout — lanes 0-15 hold K{0..7,16..23},
    // lanes 16-31 hold K{8..15,24..31}, row M = lane&15.
    V16 af[2];
#pragma unroll
    for (int i = 0; i < 2; ++i) {
      const _Float16* rp = &As[buf][wm * 32 + i * 16 + l16][hs * 8];
      af[i].h[0] = *(const v8h*)rp;
      af[i].h[1] = *(const v8h*)(rp + 16);
    }
    // B fragments: col = lane&15, K{0..7,16..23} (lanes 0-15) / K{8..15,24..31} (lanes 16-31)
    V16 bf[4];
#pragma unroll
    for (int j = 0; j < 4; ++j) {
      const _Float16* cp = &Bs[buf][wn * 64 + j * 16 + l16][hs * 8];
      bf[j].h[0] = *(const v8h*)cp;
      bf[j].h[1] = *(const v8h*)(cp + 16);
    }

#pragma unroll
    for (int i = 0; i < 2; ++i)
#pragma unroll
      for (int j = 0; j < 4; ++j)
        acc[i][j] = WMMA_F16(af[i].v, bf[j].v, acc[i][j]);

    if (kt + 1 < KT) waitTiles();
    __syncthreads();
    buf ^= 1;
  }

  // Epilogue: C/D layout — VGPR vr holds row (vr + 8*hs), col = lane&15.
#pragma unroll
  for (int i = 0; i < 2; ++i) {
    const int rbase = bM + wm * 32 + i * 16 + hs * 8;
#pragma unroll
    for (int j = 0; j < 4; ++j) {
      const int cg = bN + wn * 64 + j * 16 + l16;
      const float bv = bias[cg];
#pragma unroll
      for (int vr = 0; vr < 8; ++vr) {
        const float val = acc[i][j][vr] + bv;
        const size_t idx = (size_t)(rbase + vr) * N + cg;
        if (OUT_HALF) ((_Float16*)Cpv)[idx] = (_Float16)val;
        else          ((float*)Cpv)[idx]    = val;
      }
    }
  }
}

// ---------------------------------------------------------------------------
// Per-position head attention (one wave32 per position p = n*4096 + t).
// ---------------------------------------------------------------------------
__global__ __launch_bounds__(256) void attn_kernel(
    const _Float16* __restrict__ qkv, _Float16* __restrict__ aout)
{
  __shared__ _Float16 Ps[8][16][18];

  const int tid  = threadIdx.x;
  const int lane = tid & 31;
  const int wave = tid >> 5;
  const int hs   = lane >> 4;
  const int l16  = lane & 15;

  const size_t p = (size_t)blockIdx.x * 8 + wave;
  const _Float16* base = qkv + p * 3072;
  const _Float16* Qr = base + (size_t)l16 * 64;
  const _Float16* Kr = base + 1024 + (size_t)l16 * 64;

  V16 qa[2], kb[2];
#pragma unroll
  for (int w = 0; w < 2; ++w) {
    const _Float16* qp = Qr + w * 32 + hs * 8;
    qa[w].h[0] = *(const v8h*)qp;
    qa[w].h[1] = *(const v8h*)(qp + 16);
    const _Float16* kp = Kr + w * 32 + hs * 8;
    kb[w].h[0] = *(const v8h*)kp;
    kb[w].h[1] = *(const v8h*)(kp + 16);
  }
  v8f s = {};
  s = WMMA_F16(qa[0].v, kb[0].v, s);
  s = WMMA_F16(qa[1].v, kb[1].v, s);

  const float scale = 0.125f;   // 1/sqrt(64)
  float r[8];
#pragma unroll
  for (int e = 0; e < 8; ++e) {
    float v = s[e] * scale;
    float m = v;
    m = fmaxf(m, __shfl_xor(m, 1, 16));
    m = fmaxf(m, __shfl_xor(m, 2, 16));
    m = fmaxf(m, __shfl_xor(m, 4, 16));
    m = fmaxf(m, __shfl_xor(m, 8, 16));
    float ex = __expf(v - m);
    float sum = ex;
    sum += __shfl_xor(sum, 1, 16);
    sum += __shfl_xor(sum, 2, 16);
    sum += __shfl_xor(sum, 4, 16);
    sum += __shfl_xor(sum, 8, 16);
    r[e] = ex / sum;
  }

#pragma unroll
  for (int e = 0; e < 8; ++e) Ps[wave][e + 8 * hs][l16] = (_Float16)r[e];
  __syncthreads();

  V16 pa;
#pragma unroll
  for (int e = 0; e < 8; ++e) {
    pa.v[e]     = Ps[wave][l16][hs * 8 + e];
    pa.v[8 + e] = (_Float16)0.f;
  }

  const _Float16* Vb = base + 2048;
  const int n   = (int)(p >> 12);
  const int t   = (int)(p & 4095);
  const int q   = t >> 4;
  const int i16 = t & 15;

#pragma unroll
  for (int j = 0; j < 4; ++j) {
    V16 vb;
#pragma unroll
    for (int e = 0; e < 16; ++e) {
      const int g = (e < 8) ? hs * 8 + e : 16 + hs * 8 + (e - 8);
      vb.v[e] = (g < 16) ? Vb[(size_t)g * 64 + j * 16 + l16] : (_Float16)0.f;
    }
    v8f o = {};
    o = WMMA_F16(pa.v, vb.v, o);
#pragma unroll
    for (int vr = 0; vr < 8; ++vr) {
      const int h = vr + 8 * hs;
      const size_t row = (size_t)n * 4096 + (size_t)h * 256 + q;
      aout[row * 1024 + (size_t)i16 * 64 + j * 16 + l16] = (_Float16)o[vr];
    }
  }
}

// ---------------------------------------------------------------------------
// Pre-pass kernels: f32 -> f16 convert, and convert+transpose for weights.
// ---------------------------------------------------------------------------
__global__ __launch_bounds__(256) void cvt_kernel(
    const float* __restrict__ in, _Float16* __restrict__ out)
{
  const size_t i = ((size_t)blockIdx.x * 256 + threadIdx.x) * 8;
  const float4 a = *(const float4*)(in + i);
  const float4 b = *(const float4*)(in + i + 4);
  v8h o;
  o[0] = (_Float16)a.x; o[1] = (_Float16)a.y; o[2] = (_Float16)a.z; o[3] = (_Float16)a.w;
  o[4] = (_Float16)b.x; o[5] = (_Float16)b.y; o[6] = (_Float16)b.z; o[7] = (_Float16)b.w;
  *(v8h*)(out + i) = o;
}

// in: K x N row-major f32; out: N x K row-major f16 (transposed)
__global__ __launch_bounds__(256) void cvtT_kernel(
    const float* __restrict__ in, _Float16* __restrict__ out, int K, int N)
{
  __shared__ float tile[32][33];
  const int k0 = blockIdx.y * 32, n0 = blockIdx.x * 32;
  const int tx = threadIdx.x & 31, ty = threadIdx.x >> 5;   // ty 0..7
#pragma unroll
  for (int i = 0; i < 32; i += 8)
    tile[ty + i][tx] = in[(size_t)(k0 + ty + i) * N + n0 + tx];
  __syncthreads();
#pragma unroll
  for (int i = 0; i < 32; i += 8)
    out[(size_t)(n0 + ty + i) * K + k0 + tx] = (_Float16)tile[tx][ty + i];
}

// ---------------------------------------------------------------------------
extern "C" void kernel_launch(void* const* d_in, const int* in_sizes, int n_in,
                              void* d_out, int out_size, void* d_ws, size_t ws_size,
                              hipStream_t stream) {
  const float* x      = (const float*)d_in[0];   // (16,4096,1024)
  const float* w_qkv  = (const float*)d_in[1];   // (1024,3072)
  const float* b_qkv  = (const float*)d_in[2];   // (3072,)
  const float* w_proj = (const float*)d_in[3];   // (1024,1024)
  const float* b_proj = (const float*)d_in[4];   // (1024,)
  float* out = (float*)d_out;
  (void)ws_size; (void)in_sizes; (void)n_in; (void)out_size;

  const int M = 16 * 4096;                       // 65536 rows
  _Float16* xh     = (_Float16*)d_ws;            // M x 1024        (128 MB)
  _Float16* wqkvT  = xh     + (size_t)M * 1024;  // 3072 x 1024     (  6 MB)
  _Float16* wprojT = wqkvT  + (size_t)3072 * 1024;   // 1024 x 1024 (  2 MB)
  _Float16* qkv_h  = wprojT + (size_t)1024 * 1024;   // M x 3072    (384 MB)
  _Float16* attn_h = qkv_h  + (size_t)M * 3072;      // M x 1024    (128 MB)

  cvt_kernel<<<(M * 1024) / (256 * 8), TH, 0, stream>>>(x, xh);
  cvtT_kernel<<<dim3(3072 / 32, 1024 / 32), TH, 0, stream>>>(w_qkv, wqkvT, 1024, 3072);
  cvtT_kernel<<<dim3(1024 / 32, 1024 / 32), TH, 0, stream>>>(w_proj, wprojT, 1024, 1024);

  dim3 g1(3072 / BN, M / BM);                    // (24, 512)
  gemm_f16_kernel<true><<<g1, TH, 0, stream>>>(xh, wqkvT, b_qkv, qkv_h, M, 3072, 1024);

  attn_kernel<<<M / 8, TH, 0, stream>>>(qkv_h, attn_h);

  dim3 g2(1024 / BN, M / BM);                    // (8, 512)
  gemm_f16_kernel<false><<<g2, TH, 0, stream>>>(attn_h, wprojT, b_proj, out, M, 1024, 1024);
}
